// Mamba2_42125039239456
// MI455X (gfx1250) — matrix operand order repaired and checked by reading, not compiled, not run on hardware
//
#include <hip/hip_runtime.h>

// ---------------------------------------------------------------------------
// Mamba-2 forward for MI455X (gfx1250), wave32.
//  - All GEMM-shaped work on v_wmma_f32_16x16x32_bf16 (bf16 in, fp32 accum).
//  - Tiles staged into LDS by the Tensor Data Mover (tensor_load_to_lds),
//    double-buffered on TENSORcnt for the big GEMMs.
//  - xBC kept in bf16 end-to-end (halves SSD-stage HBM traffic).
//  - Pointwise work (conv1d, softplus, decays, gated RMSNorm) in fp32 VALU.
// ---------------------------------------------------------------------------

#define B_SZ 2
#define SEQ 4096
#define D_MODEL 1024
#define INTER 2048
#define HEADDIM 64
#define D_STATE 128
#define NHEADS 32
#define CONV_DIM 2304      // INTER + 2*D_STATE
#define PROJ_DIM 4385      // 2*INTER + 2*D_STATE + NHEADS + 1
#define PROJ_PAD 4416      // next multiple of 64
#define CHUNK 256
#define NCHUNK 16          // SEQ / CHUNK
#define ROWS 8192          // B_SZ * SEQ

typedef __attribute__((ext_vector_type(16))) __bf16 v16bf;
typedef __attribute__((ext_vector_type(8)))  float  v8f;
typedef __attribute__((ext_vector_type(4)))  unsigned int u32x4;
typedef __attribute__((ext_vector_type(8)))  unsigned int u32x8;

union BF16x16 { unsigned short u[16]; v16bf v; };

__device__ __forceinline__ unsigned short f2bf(float f) {
  unsigned int x = __float_as_uint(f);
  x += 0x7FFFu + ((x >> 16) & 1u);     // round-to-nearest-even
  return (unsigned short)(x >> 16);
}
__device__ __forceinline__ float bf2f(unsigned short h) {
  return __uint_as_float(((unsigned int)h) << 16);
}

__device__ __forceinline__ v8f wmma_bf16(v16bf a, v16bf b, v8f c) {
  return __builtin_amdgcn_wmma_f32_16x16x32_bf16(false, a, false, b,
                                                 (short)0, c, false, false);
}

// ---- fragment loaders (k = 16*half + j in A and B) --------------------------------
// A 16x32 bf16, source stored row-major [M][K]
__device__ __forceinline__ v16bf load_a_row(const unsigned short* A, int lda,
                                            int m0, int k0, int lane) {
  BF16x16 r;
  int m = m0 + (lane & 15);
  int half = lane >> 4;
  const unsigned short* p0 = A + (size_t)m * lda + k0 + 16 * half;
#pragma unroll
  for (int j = 0; j < 16; ++j) r.u[j] = p0[j];
  return r.v;
}
// A 16x32 bf16, source stored [K][M] (transposed read)
__device__ __forceinline__ v16bf load_a_col(const unsigned short* A, int lda,
                                            int m0, int k0, int lane) {
  BF16x16 r;
  int m = m0 + (lane & 15);
  int half = lane >> 4;
#pragma unroll
  for (int j = 0; j < 16; ++j)
    r.u[j] = A[(size_t)(k0 + 16 * half + j) * lda + m];
  return r.v;
}
// B 32x16 bf16, source stored [N][K] (B^T row-major): contiguous per lane
__device__ __forceinline__ v16bf load_b_nk(const unsigned short* B, int ldb,
                                           int n0, int k0, int lane) {
  BF16x16 r;
  int n = n0 + (lane & 15);
  int half = lane >> 4;
  const unsigned short* p = B + (size_t)n * ldb + k0 + 16 * half;
#pragma unroll
  for (int j = 0; j < 16; ++j) r.u[j] = p[j];
  return r.v;
}
// B 32x16 bf16, source stored [K][N] row-major
__device__ __forceinline__ v16bf load_b_kn(const unsigned short* B, int ldb,
                                           int n0, int k0, int lane) {
  BF16x16 r;
  int n = n0 + (lane & 15);
  int half = lane >> 4;
#pragma unroll
  for (int j = 0; j < 16; ++j)
    r.u[j] = B[(size_t)(k0 + 16 * half + j) * ldb + n];
  return r.v;
}

// ---------------------------------------------------------------------------
// TDM: 2D bf16 tile -> LDS.  Descriptor per ISA ch.8 (group0 + group1).
// lds_base is passed as an asm operand so the shared object ESCAPES -> the
// "memory" clobber makes the TDM's LDS writes visible to later ds_loads.
// ---------------------------------------------------------------------------
__device__ __forceinline__ void tdm_load_2d(unsigned lds_off, void* lds_base,
                                            const void* gptr,
                                            unsigned tile_k, unsigned tile_rows,
                                            unsigned long long stride_elems) {
  unsigned long long ga = (unsigned long long)(size_t)gptr;
  u32x4 g0;
  g0[0] = 1u;                                          // count=1, no gather
  g0[1] = lds_off;                                     // LDS byte address
  g0[2] = (unsigned)(ga & 0xffffffffull);              // global_addr[31:0]
  g0[3] = (unsigned)((ga >> 32) & 0x1ffffffull)        // global_addr[56:32]
        | (2u << 30);                                  // type = 2 ("image")
  const unsigned td = 0x7fffffffu;                     // generous dims (in-bounds tiles)
  u32x8 g1;
  g1[0] = (1u << 16);                                  // data_size = 2 bytes
  g1[1] = (td & 0xffffu) << 16;                        // tensor_dim0[15:0]
  g1[2] = ((td >> 16) & 0xffffu)                       // tensor_dim0[31:16]
        | ((td & 0xffffu) << 16);                      // tensor_dim1[15:0]
  g1[3] = ((td >> 16) & 0xffffu)                       // tensor_dim1[31:16]
        | ((tile_k & 0xffffu) << 16);                  // tile_dim0
  g1[4] = (tile_rows & 0xffffu);                       // tile_dim1 (tile_dim2=0)
  g1[5] = (unsigned)(stride_elems & 0xffffffffull);    // dim0 stride[31:0]
  g1[6] = (unsigned)((stride_elems >> 32) & 0xffffull);// dim0 stride[47:32]
  g1[7] = 0u;
  asm volatile("tensor_load_to_lds %0, %1"
               :: "s"(g0), "s"(g1), "s"(lds_base) : "memory");
}

// ---------------------------------------------------------------------------
// bf16 GEMM: C[M x Nreal] = A[M x K] * Bt[Npad x K]^T, fp32 out.
// Block = 8 waves; block tile 128(M) x 64(N); K staged 64 at a time through
// LDS by the TDM, double-buffered (TENSORcnt <= 2 overlap).  All 10 fragments
// of a stage are loaded into VGPRs first, then 8 WMMAs issue back-to-back.
// LDS struct is the only shared allocation -> starts at LDS offset 0.
// ---------------------------------------------------------------------------
struct GemmLds {
  unsigned short A[2][128 * 64];   // 2 x 16 KB  (offsets 0, 16384)
  unsigned short B[2][64 * 64];    // 2 x  8 KB  (offsets 32768, 40960)
};
#define LDS_A(buf) ((buf) ? 16384u : 0u)
#define LDS_B(buf) (32768u + ((buf) ? 8192u : 0u))

__global__ __launch_bounds__(256) void k_gemm_bf16(
    const unsigned short* __restrict__ A, const unsigned short* __restrict__ Bt,
    float* __restrict__ C, int M, int K, int Nreal) {
  __shared__ GemmLds lds;
  int lane = threadIdx.x & 31;
  int wave = threadIdx.x >> 5;
  int mBlk = blockIdx.y * 128;
  int nBlk = blockIdx.x * 64;
  int nstage = K >> 6;

  if (threadIdx.x == 0) {
    tdm_load_2d(LDS_A(0), &lds, A + (size_t)mBlk * K, 64, 128,
                (unsigned long long)K);
    tdm_load_2d(LDS_B(0), &lds, Bt + (size_t)nBlk * K, 64, 64,
                (unsigned long long)K);
  }

  v8f acc[4] = {};
  for (int s = 0; s < nstage; ++s) {
    int buf = s & 1;
    if (threadIdx.x == 0) {
      if (s + 1 < nstage) {
        int k0 = (s + 1) << 6;
        tdm_load_2d(LDS_A(buf ^ 1), &lds, A + (size_t)mBlk * K + k0, 64, 128,
                    (unsigned long long)K);
        tdm_load_2d(LDS_B(buf ^ 1), &lds, Bt + (size_t)nBlk * K + k0, 64, 64,
                    (unsigned long long)K);
        __builtin_amdgcn_s_wait_tensorcnt(2);   // stage s landed, s+1 in flight
      } else {
        __builtin_amdgcn_s_wait_tensorcnt(0);
      }
    }
    __syncthreads();
    const unsigned short* At = lds.A[buf];
    const unsigned short* Bl = lds.B[buf];
    // batch all fragment loads, then issue WMMAs back-to-back
    v16bf a0 = load_a_row(At, 64, wave * 16, 0, lane);
    v16bf a1 = load_a_row(At, 64, wave * 16, 32, lane);
    v16bf b0[4], b1[4];
#pragma unroll
    for (int t = 0; t < 4; ++t) {
      b0[t] = load_b_nk(Bl, 64, t * 16, 0, lane);
      b1[t] = load_b_nk(Bl, 64, t * 16, 32, lane);
    }
#pragma unroll
    for (int t = 0; t < 4; ++t) acc[t] = wmma_bf16(a0, b0[t], acc[t]);
#pragma unroll
    for (int t = 0; t < 4; ++t) acc[t] = wmma_bf16(a1, b1[t], acc[t]);
    __syncthreads();
  }

  int half = lane >> 4, nl = lane & 15;
#pragma unroll
  for (int t = 0; t < 4; ++t) {
    int n = nBlk + t * 16 + nl;
    if (n < Nreal) {
#pragma unroll
      for (int r = 0; r < 8; ++r) {
        int m = mBlk + wave * 16 + r + 8 * half;
        C[(size_t)m * Nreal + n] = acc[t][r];
      }
    }
  }
}

// ---- conversion helpers ---------------------------------------------------
__global__ void k_f32_to_bf16(const float* __restrict__ in,
                              unsigned short* __restrict__ out, int n) {
  int i = blockIdx.x * 256 + threadIdx.x;
  if (i < n) out[i] = f2bf(in[i]);
}
// W [K x N] fp32 -> Wt [Npad x K] bf16 (zero pad)
__global__ void k_transpose_bf16(const float* __restrict__ W,
                                 unsigned short* __restrict__ Wt,
                                 int K, int N, int Npad) {
  int i = blockIdx.x * 256 + threadIdx.x;
  if (i >= Npad * K) return;
  int n = i / K, k = i - n * K;
  Wt[i] = (n < N) ? f2bf(W[(size_t)k * N + n]) : (unsigned short)0;
}

// ---- causal conv1d (k=4) + SiLU -> bf16 xBC -------------------------------
__global__ void k_conv_silu(const float* __restrict__ zx,
                            const float* __restrict__ cw,
                            const float* __restrict__ cb,
                            unsigned short* __restrict__ xBC16) {
  int i = blockIdx.x * 256 + threadIdx.x;
  if (i >= B_SZ * SEQ * CONV_DIM) return;
  int c = i % CONV_DIM;
  int s = (i / CONV_DIM) % SEQ;
  int b = i / (CONV_DIM * SEQ);
  float acc = cb[c];
#pragma unroll
  for (int t = 0; t < 4; ++t) {
    int sp = s - 3 + t;
    if (sp >= 0)
      acc += zx[(size_t)(b * SEQ + sp) * PROJ_DIM + INTER + c] * cw[c * 4 + t];
  }
  xBC16[i] = f2bf(acc / (1.f + expf(-acc)));   // silu
}

// ---- dt = softplus(dt_raw + bias) ----------------------------------------
__global__ void k_dt(const float* __restrict__ zx,
                     const float* __restrict__ dt_bias,
                     float* __restrict__ dtb) {
  int i = blockIdx.x * 256 + threadIdx.x;
  if (i >= ROWS * NHEADS) return;
  int h = i % NHEADS, row = i / NHEADS;
  float x = zx[(size_t)row * PROJ_DIM + INTER + CONV_DIM + h] + dt_bias[h];
  dtb[i] = (x > 20.f) ? x : log1pf(expf(x));
}

// ---------------------------------------------------------------------------
// SSD intra-chunk kernel: one block per (head, chunk, batch).
// Bc/Cc/X staged by TDM (overlapped with the decay scan); X scaled by dt in
// LDS.  LDS: Bc/Cc [256x128] bf16, Xd [256x64] bf16, Acum/dts f32,
// SL [256x256] bf16 (reused as Xdec).
// ---------------------------------------------------------------------------
#define SMEM_BC   0
#define SMEM_CC   65536
#define SMEM_XD   131072
#define SMEM_ACUM 163840
#define SMEM_DTS  164864
#define SMEM_SL   165888
#define SMEM_SSD  (165888 + 131072)   // 296960 bytes < 320KB WGP LDS

__global__ __launch_bounds__(256) void k_ssd_chunk(
    const unsigned short* __restrict__ xBC16, const float* __restrict__ dtb,
    const float* __restrict__ A_log, const float* __restrict__ D_param,
    float* __restrict__ y, float* __restrict__ states,
    float* __restrict__ Acumg) {
  extern __shared__ char smem[];
  unsigned short* Bc   = (unsigned short*)(smem + SMEM_BC);
  unsigned short* Cc   = (unsigned short*)(smem + SMEM_CC);
  unsigned short* Xd   = (unsigned short*)(smem + SMEM_XD);
  float*          Acum = (float*)(smem + SMEM_ACUM);
  float*          dts  = (float*)(smem + SMEM_DTS);
  unsigned short* SL   = (unsigned short*)(smem + SMEM_SL);
  unsigned short* Xdec = SL;   // reused after Y_diag

  int h = blockIdx.x, c = blockIdx.y, b = blockIdx.z;
  int tid = threadIdx.x, lane = tid & 31, wave = tid >> 5;
  size_t rowbase = (size_t)b * SEQ + (size_t)c * CHUNK;
  float Ah = -expf(A_log[h]);

  // kick off the three tile DMAs; they overlap the decay scan below
  const unsigned short* xrow = xBC16 + rowbase * CONV_DIM;
  if (tid == 0) {
    tdm_load_2d(SMEM_BC, smem, xrow + INTER, 128, 256, CONV_DIM);
    tdm_load_2d(SMEM_CC, smem, xrow + INTER + D_STATE, 128, 256, CONV_DIM);
    tdm_load_2d(SMEM_XD, smem, xrow + h * HEADDIM, 64, 256, CONV_DIM);
  }

  int l = tid;
  float dtv = dtb[(rowbase + l) * NHEADS + h];
  dts[l] = dtv;
  Acum[l] = dtv * Ah;
  __syncthreads();
  for (int off = 1; off < 256; off <<= 1) {
    float add = (l >= off) ? Acum[l - off] : 0.f;
    __syncthreads();
    Acum[l] += add;
    __syncthreads();
  }
  size_t abase = (((size_t)b * NCHUNK + c) * NHEADS + h) * 256;
  Acumg[abase + l] = Acum[l];

  if (tid == 0) __builtin_amdgcn_s_wait_tensorcnt(0);
  __syncthreads();

  // scale X tile by dt in place (LDS rmw)
  for (int t = tid; t < 256 * 64; t += 256) {
    int lr = t >> 6;
    Xd[t] = f2bf(bf2f(Xd[t]) * dts[lr]);
  }
  __syncthreads();

  // phase 1: SL = (Cc @ Bc^T) * L  (skip strictly-upper causal tiles)
  for (int tile = wave; tile < 256; tile += 8) {
    int ti = tile >> 4, tj = tile & 15;
    v8f acc = {};
    if (tj <= ti) {
#pragma unroll
      for (int k0 = 0; k0 < 128; k0 += 32) {
        v16bf a  = load_a_row(Cc, 128, ti * 16, k0, lane);
        v16bf bb = load_b_nk(Bc, 128, tj * 16, k0, lane);
        acc = wmma_bf16(a, bb, acc);
      }
    }
    int half = lane >> 4, sl = lane & 15;
    int scol = tj * 16 + sl;
#pragma unroll
    for (int r = 0; r < 8; ++r) {
      int lrow = ti * 16 + r + 8 * half;
      float v = (scol <= lrow) ? acc[r] * expf(Acum[lrow] - Acum[scol]) : 0.f;
      SL[lrow * 256 + scol] = f2bf(v);
    }
  }
  __syncthreads();

  // phase 2: Y_diag = SL @ Xd  (+ D * x skip term)
  float Dh = D_param[h];
  for (int tile = wave; tile < 64; tile += 8) {
    int ti = tile >> 2, tj = tile & 3;
    v8f acc = {};
#pragma unroll
    for (int k0 = 0; k0 < 256; k0 += 32) {
      v16bf a  = load_a_row(SL, 256, ti * 16, k0, lane);
      v16bf bb = load_b_kn(Xd, 64, tj * 16, k0, lane);
      acc = wmma_bf16(a, bb, acc);
    }
    int half = lane >> 4, pl = lane & 15;
    int p = tj * 16 + pl;
#pragma unroll
    for (int r = 0; r < 8; ++r) {
      int lrow = ti * 16 + r + 8 * half;
      size_t row = rowbase + lrow;
      float xv = bf2f(xBC16[row * CONV_DIM + h * HEADDIM + p]);
      y[row * INTER + h * HEADDIM + p] = acc[r] + Dh * xv;
    }
  }
  __syncthreads();

  // Xdec = Xd * exp(Acum_last - Acum[l])   (reuses SL LDS region)
  float Alast = Acum[255];
  for (int t = tid; t < 256 * 64; t += 256) {
    int lr = t >> 6;
    Xdec[t] = f2bf(bf2f(Xd[t]) * expf(Alast - Acum[lr]));
  }
  __syncthreads();

  // phase 3: states[n][p] = Bc^T @ Xdec   (128 x 64, K = 256)
  size_t sbase = (((size_t)b * NCHUNK + c) * NHEADS + h) * (size_t)(D_STATE * HEADDIM);
  for (int tile = wave; tile < 32; tile += 8) {
    int ti = tile >> 2, tj = tile & 3;
    v8f acc = {};
#pragma unroll
    for (int k0 = 0; k0 < 256; k0 += 32) {
      v16bf a  = load_a_col(Bc, 128, ti * 16, k0, lane);   // Bc^T
      v16bf bb = load_b_kn(Xdec, 64, tj * 16, k0, lane);
      acc = wmma_bf16(a, bb, acc);
    }
    int half = lane >> 4, pl = lane & 15;
    int p = tj * 16 + pl;
#pragma unroll
    for (int r = 0; r < 8; ++r) {
      int n = ti * 16 + r + 8 * half;
      states[sbase + (size_t)n * HEADDIM + p] = acc[r];
    }
  }
}

// ---- inter-chunk sequential state scan ------------------------------------
__global__ __launch_bounds__(256) void k_ssd_scan(
    const float* __restrict__ states, const float* __restrict__ Acumg,
    float* __restrict__ prior) {
  int h = blockIdx.x, b = blockIdx.y;
  int tid = threadIdx.x;
  float run[32];
#pragma unroll
  for (int j = 0; j < 32; ++j) run[j] = 0.f;
  for (int c = 0; c < NCHUNK; ++c) {
    size_t base = (((size_t)b * NCHUNK + c) * NHEADS + h) * 8192 + (size_t)tid * 32;
#pragma unroll
    for (int j = 0; j < 32; ++j) prior[base + j] = run[j];
    float dec = expf(Acumg[(((size_t)b * NCHUNK + c) * NHEADS + h) * 256 + 255]);
#pragma unroll
    for (int j = 0; j < 32; ++j) run[j] = run[j] * dec + states[base + j];
  }
}

// ---- Y_off = (Cc * exp(Acum)) @ prior_state -------------------------------
__global__ __launch_bounds__(256) void k_ssd_yoff(
    const unsigned short* __restrict__ xBC16, const float* __restrict__ Acumg,
    const float* __restrict__ prior, float* __restrict__ y) {
  extern __shared__ char smem[];
  unsigned short* CcE = (unsigned short*)smem;             // [256][128]
  unsigned short* Ps  = (unsigned short*)(smem + 65536);   // [128][64]
  int h = blockIdx.x, c = blockIdx.y, b = blockIdx.z;
  int tid = threadIdx.x, lane = tid & 31, wave = tid >> 5;
  size_t rowbase = (size_t)b * SEQ + (size_t)c * CHUNK;
  size_t abase = (((size_t)b * NCHUNK + c) * NHEADS + h) * 256;

  if (tid == 0)
    tdm_load_2d(0, smem, xBC16 + rowbase * CONV_DIM + INTER + D_STATE,
                128, 256, CONV_DIM);

  size_t pbase = (((size_t)b * NCHUNK + c) * NHEADS + h) * 8192;
  for (int t = tid; t < 128 * 64; t += 256) Ps[t] = f2bf(prior[pbase + t]);

  if (tid == 0) __builtin_amdgcn_s_wait_tensorcnt(0);
  __syncthreads();
  // scale C rows by exp(Acum) in place
  for (int t = tid; t < 256 * 128; t += 256) {
    int lr = t >> 7;
    CcE[t] = f2bf(bf2f(CcE[t]) * expf(Acumg[abase + lr]));
  }
  __syncthreads();

  for (int tile = wave; tile < 64; tile += 8) {
    int ti = tile >> 2, tj = tile & 3;
    v8f acc = {};
#pragma unroll
    for (int k0 = 0; k0 < 128; k0 += 32) {
      v16bf a  = load_a_row(CcE, 128, ti * 16, k0, lane);
      v16bf bb = load_b_kn(Ps, 64, tj * 16, k0, lane);
      acc = wmma_bf16(a, bb, acc);
    }
    int half = lane >> 4, pl = lane & 15;
    int p = tj * 16 + pl;
#pragma unroll
    for (int r = 0; r < 8; ++r) {
      int lrow = ti * 16 + r + 8 * half;
      size_t row = rowbase + lrow;
      y[row * INTER + h * HEADDIM + p] += acc[r];
    }
  }
}

// ---- gated RMSNorm -> bf16 ------------------------------------------------
__global__ __launch_bounds__(256) void k_gatenorm(
    const float* __restrict__ y, const float* __restrict__ zx,
    const float* __restrict__ norm_w, unsigned short* __restrict__ g) {
  __shared__ float red[256];
  __shared__ float gs[INTER];
  int row = blockIdx.x, tid = threadIdx.x;
  const float* zrow = zx + (size_t)row * PROJ_DIM;   // z slice [0..INTER)
  const float* yrow = y + (size_t)row * INTER;
  float ssum = 0.f;
  for (int i = tid; i < INTER; i += 256) {
    float z = zrow[i];
    float gv = yrow[i] * (z / (1.f + expf(-z)));
    gs[i] = gv;
    ssum += gv * gv;
  }
  red[tid] = ssum;
  __syncthreads();
  for (int off = 128; off > 0; off >>= 1) {
    if (tid < off) red[tid] += red[tid + off];
    __syncthreads();
  }
  float scale = rsqrtf(red[0] * (1.f / INTER) + 1e-5f);
  for (int i = tid; i < INTER; i += 256)
    g[(size_t)row * INTER + i] = f2bf(gs[i] * scale * norm_w[i]);
}

// ---------------------------------------------------------------------------
extern "C" void kernel_launch(void* const* d_in, const int* in_sizes, int n_in,
                              void* d_out, int out_size, void* d_ws, size_t ws_size,
                              hipStream_t stream) {
  const float* u        = (const float*)d_in[0];
  const float* in_w     = (const float*)d_in[1];
  const float* conv_w   = (const float*)d_in[2];
  const float* conv_b   = (const float*)d_in[3];
  const float* dt_bias  = (const float*)d_in[4];
  const float* A_log    = (const float*)d_in[5];
  const float* D_param  = (const float*)d_in[6];
  const float* norm_w   = (const float*)d_in[7];
  const float* out_w    = (const float*)d_in[8];

  char* ws = (char*)d_ws;
  size_t o = 0;
  unsigned short* WtIn  = (unsigned short*)(ws + o); o += (size_t)PROJ_PAD * D_MODEL * 2;
  unsigned short* WtOut = (unsigned short*)(ws + o); o += (size_t)D_MODEL * INTER * 2;
  unsigned short* u16   = (unsigned short*)(ws + o); o += (size_t)ROWS * D_MODEL * 2;
  float* zx     = (float*)(ws + o); o += (size_t)ROWS * PROJ_DIM * 4;
  unsigned short* xBC16 = (unsigned short*)(ws + o); o += (size_t)ROWS * CONV_DIM * 2;
  float* dtb    = (float*)(ws + o); o += (size_t)ROWS * NHEADS * 4;
  float* ybuf   = (float*)(ws + o); o += (size_t)ROWS * INTER * 4;
  float* states = (float*)(ws + o); o += (size_t)B_SZ * NCHUNK * NHEADS * D_STATE * HEADDIM * 4;
  float* prior  = (float*)(ws + o); o += (size_t)B_SZ * NCHUNK * NHEADS * D_STATE * HEADDIM * 4;
  float* Acumg  = (float*)(ws + o); o += (size_t)B_SZ * NCHUNK * NHEADS * 256 * 4;
  unsigned short* g16 = (unsigned short*)(ws + o); o += (size_t)ROWS * INTER * 2;

  // 1. weight / activation conversion to bf16 (weights transposed to [N][K])
  k_transpose_bf16<<<(PROJ_PAD * D_MODEL + 255) / 256, 256, 0, stream>>>(
      in_w, WtIn, D_MODEL, PROJ_DIM, PROJ_PAD);
  k_transpose_bf16<<<(D_MODEL * INTER + 255) / 256, 256, 0, stream>>>(
      out_w, WtOut, INTER, D_MODEL, D_MODEL);
  k_f32_to_bf16<<<(ROWS * D_MODEL + 255) / 256, 256, 0, stream>>>(
      u, u16, ROWS * D_MODEL);

  // 2. in_proj GEMM: [8192 x 1024] x [1024 x 4385] (TDM-staged, WMMA)
  k_gemm_bf16<<<dim3(PROJ_PAD / 64, ROWS / 128), 256, 0, stream>>>(
      u16, WtIn, zx, ROWS, D_MODEL, PROJ_DIM);

  // 3. causal conv + silu (-> bf16), dt softplus
  k_conv_silu<<<(B_SZ * SEQ * CONV_DIM + 255) / 256, 256, 0, stream>>>(
      zx, conv_w, conv_b, xBC16);
  k_dt<<<(ROWS * NHEADS + 255) / 256, 256, 0, stream>>>(zx, dt_bias, dtb);

  // 4. SSD intra-chunk (TDM + WMMA), inter-chunk scan, off-diagonal (WMMA)
  k_ssd_chunk<<<dim3(NHEADS, NCHUNK, B_SZ), 256, SMEM_SSD, stream>>>(
      xBC16, dtb, A_log, D_param, ybuf, states, Acumg);
  k_ssd_scan<<<dim3(NHEADS, B_SZ), 256, 0, stream>>>(states, Acumg, prior);
  k_ssd_yoff<<<dim3(NHEADS, NCHUNK, B_SZ), 256, 65536 + 16384, stream>>>(
      xBC16, Acumg, prior, ybuf);

  // 5. gated RMSNorm -> bf16, out_proj GEMM into d_out (TDM-staged, WMMA)
  k_gatenorm<<<ROWS, 256, 0, stream>>>(ybuf, zx, norm_w, g16);
  k_gemm_bf16<<<dim3(D_MODEL / 64, ROWS / 128), 256, 0, stream>>>(
      g16, WtOut, (float*)d_out, ROWS, INTER, D_MODEL);
}
